// SpeakerStateRNN_83099027243215
// MI455X (gfx1250) — matrix operand order repaired and checked by reading, not compiled, not run on hardware
//
#include <hip/hip_runtime.h>
#include <hip/hip_bf16.h>

#define Bz 128
#define Tz 256
#define Dz 768
#define Hz 512
#define N3 1536  // 3*Hz

typedef __attribute__((ext_vector_type(16))) __bf16 v16bf;
typedef __attribute__((ext_vector_type(8)))  float  v8f;
typedef __attribute__((ext_vector_type(4)))  float  v4f;

static __device__ __forceinline__ v8f wmma_bf16(v16bf a, v16bf b, v8f c) {
    // D = A(16x32 bf16) * B(32x16 bf16) + C(16x16 f32)
    return __builtin_amdgcn_wmma_f32_16x16x32_bf16(false, a, false, b, (short)0, c,
                                                   false, false);
}

static __device__ __forceinline__ float sigmoidf_(float x) {
    return 1.0f / (1.0f + __expf(-x));
}
static __device__ __forceinline__ float tanhf_(float x) {
    return 2.0f / (1.0f + __expf(-2.0f * x)) - 1.0f;
}

// Load 8 contiguous f32 from p, convert to bf16 into elements [off..off+7] of v.
static __device__ __forceinline__ void load8_cvt(v16bf& v, int off,
                                                 const float* __restrict__ p) {
    v4f a = *(const v4f*)(p);
    v4f b = *(const v4f*)(p + 4);
#pragma unroll
    for (int i = 0; i < 4; ++i) {
        v[off + i]     = (__bf16)a[i];
        v[off + 4 + i] = (__bf16)b[i];
    }
}

// Build A-fragment (16x32 bf16) for this lane.
// ISA layout: lanes 0-15 = rows M, hi=lane>>4 selects K half:
//   elements 0..7  -> K = kb + 8*hi + e
//   elements 8..15 -> K = kb + 16 + 8*hi + (e-8)
// Row data is piecewise (concat input): k in [0,len0)->row0[k], else row1[k-len0].
// len0 is a multiple of 8 and chunks are 8-aligned, so chunks never straddle.
static __device__ __forceinline__ v16bf load_a(const float* __restrict__ row0, int len0,
                                               const float* __restrict__ row1,
                                               int kb, int hi) {
    int k0 = kb + 8 * hi;
    int k1 = kb + 16 + 8 * hi;
    const float* p0 = (k0 < len0) ? (row0 + k0) : (row1 + (k0 - len0));
    const float* p1 = (k1 < len0) ? (row0 + k1) : (row1 + (k1 - len0));
    v16bf a;
    load8_cvt(a, 0, p0);
    load8_cvt(a, 8, p1);
    return a;
}

// Accumulate 3 gate tiles for TWO M-tiles sharing each B-fragment:
//   accA[g] += A0 * B(g), accB[g] += A1 * B(g)
// Wt is bf16 [K x N3] row-major; B-fragment: lane l holds column N=nBase+(l&15),
// 16 bf16 gathered over K in the same order as A. Sharing B across two A frags
// halves L2 weight traffic per step.
static __device__ __forceinline__ void gemm3x2(v8f accA[3], v8f accB[3],
                                               const float* __restrict__ r0a, int len0,
                                               const float* __restrict__ r1a,
                                               const float* __restrict__ r0b,
                                               const float* __restrict__ r1b,
                                               int K, const __bf16* __restrict__ wt,
                                               int nBase, int lane, int hi) {
    for (int kb = 0; kb < K; kb += 32) {
        v16bf a0 = load_a(r0a, len0, r1a, kb, hi);
        v16bf a1 = load_a(r0b, len0, r1b, kb, hi);
        const __bf16* wrow = wt + (size_t)kb * N3 + nBase + (lane & 15);
#pragma unroll
        for (int g = 0; g < 3; ++g) {
            v16bf b;
#pragma unroll
            for (int s = 0; s < 16; ++s) {
                int kk = (s < 8) ? (8 * hi + s) : (16 + 8 * hi + (s - 8));
                b[s] = wrow[(size_t)kk * N3 + g * Hz];
            }
            accA[g] = wmma_bf16(a0, b, accA[g]);
            accB[g] = wmma_bf16(a1, b, accB[g]);
        }
    }
}

// -------- GRU cell kernels: one wave owns 2 M-tiles x 1 N-tile x 3 gates --------
// grid = 32 blocks * 4 waves = 128 waves = (B/32=4 M-pairs) x (H/16=32 N-tiles)

__global__ void __launch_bounds__(128)
k_global_cell(const float* __restrict__ utt, int t,
              const float* __restrict__ hin,
              const __bf16* __restrict__ Wih, const __bf16* __restrict__ Whh,
              const float* __restrict__ bih, const float* __restrict__ bhh,
              float* __restrict__ hout) {
    int lane = threadIdx.x & 31, wave = threadIdx.x >> 5;
    int tile = blockIdx.x * 4 + wave;
    int nBase = (tile & 31) * 16;
    int mBase0 = (tile >> 5) * 32, mBase1 = mBase0 + 16;
    int hi = lane >> 4, lr = lane & 15;

    int bA0 = mBase0 + lr, bA1 = mBase1 + lr;  // this lane's A rows (batch idx)
    const float* x0 = utt + ((size_t)bA0 * Tz + t) * Dz;
    const float* x1 = utt + ((size_t)bA1 * Tz + t) * Dz;
    const float* h0 = hin + (size_t)bA0 * Hz;
    const float* h1 = hin + (size_t)bA1 * Hz;

    v8f aci0[3] = {{0}}, aci1[3] = {{0}}, ach0[3] = {{0}}, ach1[3] = {{0}};
    gemm3x2(aci0, aci1, x0, Dz, nullptr, x1, nullptr, Dz, Wih, nBase, lane, hi);
    gemm3x2(ach0, ach1, h0, Hz, nullptr, h1, nullptr, Hz, Whh, nBase, lane, hi);

    int col = nBase + lr;
    float br = bih[col] + bhh[col];
    float bz = bih[Hz + col] + bhh[Hz + col];
    float bi_n = bih[2 * Hz + col], bh_n = bhh[2 * Hz + col];
#pragma unroll
    for (int p = 0; p < 2; ++p) {
        v8f* ai = p ? aci1 : aci0;
        v8f* ah = p ? ach1 : ach0;
        int mb = p ? mBase1 : mBase0;
#pragma unroll
        for (int v = 0; v < 8; ++v) {
            int m = mb + v + 8 * hi;  // C/D layout: M = v + 8*(lane>>4), N = lane&15
            float r  = sigmoidf_(ai[0][v] + ah[0][v] + br);
            float z  = sigmoidf_(ai[1][v] + ah[1][v] + bz);
            float nn = tanhf_(ai[2][v] + bi_n + r * (ah[2][v] + bh_n));
            float hp = hin[(size_t)m * Hz + col];
            hout[(size_t)m * Hz + col] = (1.0f - z) * nn + z * hp;
        }
    }
}

__global__ void __launch_bounds__(128)
k_speaker_cell(const float* __restrict__ utt, int t,
               const float* __restrict__ hg,
               const int* __restrict__ sid,
               const float* __restrict__ spkStates,
               const __bf16* __restrict__ Wih, const __bf16* __restrict__ Whh,
               const float* __restrict__ bih, const float* __restrict__ bhh,
               float* __restrict__ spkNew) {
    int lane = threadIdx.x & 31, wave = threadIdx.x >> 5;
    int tile = blockIdx.x * 4 + wave;
    int nBase = (tile & 31) * 16;
    int mBase0 = (tile >> 5) * 32, mBase1 = mBase0 + 16;
    int hi = lane >> 4, lr = lane & 15;

    int bA0 = mBase0 + lr, bA1 = mBase1 + lr;
    const float* x0a = utt + ((size_t)bA0 * Tz + t) * Dz;  // k in [0,768)
    const float* x0b = utt + ((size_t)bA1 * Tz + t) * Dz;
    const float* x1a = hg + (size_t)bA0 * Hz;              // k in [768,1280)
    const float* x1b = hg + (size_t)bA1 * Hz;
    int spk0 = sid[(size_t)bA0 * Tz + t];
    int spk1 = sid[(size_t)bA1 * Tz + t];
    const float* h0 = spkStates + ((size_t)bA0 * 2 + spk0) * Hz;  // per-row gather
    const float* h1 = spkStates + ((size_t)bA1 * 2 + spk1) * Hz;

    v8f aci0[3] = {{0}}, aci1[3] = {{0}}, ach0[3] = {{0}}, ach1[3] = {{0}};
    gemm3x2(aci0, aci1, x0a, Dz, x1a, x0b, x1b, Dz + Hz, Wih, nBase, lane, hi);
    gemm3x2(ach0, ach1, h0, Hz, nullptr, h1, nullptr, Hz, Whh, nBase, lane, hi);

    int col = nBase + lr;
    float br = bih[col] + bhh[col];
    float bz = bih[Hz + col] + bhh[Hz + col];
    float bi_n = bih[2 * Hz + col], bh_n = bhh[2 * Hz + col];
#pragma unroll
    for (int p = 0; p < 2; ++p) {
        v8f* ai = p ? aci1 : aci0;
        v8f* ah = p ? ach1 : ach0;
        int mb = p ? mBase1 : mBase0;
#pragma unroll
        for (int v = 0; v < 8; ++v) {
            int m = mb + v + 8 * hi;
            int spk = sid[(size_t)m * Tz + t];
            float r  = sigmoidf_(ai[0][v] + ah[0][v] + br);
            float z  = sigmoidf_(ai[1][v] + ah[1][v] + bz);
            float nn = tanhf_(ai[2][v] + bi_n + r * (ah[2][v] + bh_n));
            float hp = spkStates[((size_t)m * 2 + spk) * Hz + col];
            spkNew[(size_t)m * Hz + col] = (1.0f - z) * nn + z * hp;
        }
    }
}

__global__ void __launch_bounds__(128)
k_emotion_cell(const float* __restrict__ utt, int t,
               const float* __restrict__ spkNew,
               const __bf16* __restrict__ Wih,
               const float* __restrict__ bih, const float* __restrict__ bhh,
               float* __restrict__ out) {
    int lane = threadIdx.x & 31, wave = threadIdx.x >> 5;
    int tile = blockIdx.x * 4 + wave;
    int nBase = (tile & 31) * 16;
    int mBase0 = (tile >> 5) * 32, mBase1 = mBase0 + 16;
    int hi = lane >> 4, lr = lane & 15;

    int bA0 = mBase0 + lr, bA1 = mBase1 + lr;
    const float* x0a = utt + ((size_t)bA0 * Tz + t) * Dz;
    const float* x0b = utt + ((size_t)bA1 * Tz + t) * Dz;
    const float* x1a = spkNew + (size_t)bA0 * Hz;
    const float* x1b = spkNew + (size_t)bA1 * Hz;

    v8f aci0[3] = {{0}}, aci1[3] = {{0}};
    gemm3x2(aci0, aci1, x0a, Dz, x1a, x0b, x1b, Dz + Hz, Wih, nBase, lane, hi);

    // h == 0 for the emotion cell: gh = b_hh only, output = (1-z)*n
    int col = nBase + lr;
    float br = bih[col] + bhh[col];
    float bz = bih[Hz + col] + bhh[Hz + col];
    float bi_n = bih[2 * Hz + col], bh_n = bhh[2 * Hz + col];
#pragma unroll
    for (int p = 0; p < 2; ++p) {
        v8f* ai = p ? aci1 : aci0;
        int mb = p ? mBase1 : mBase0;
#pragma unroll
        for (int v = 0; v < 8; ++v) {
            int m = mb + v + 8 * hi;
            float r  = sigmoidf_(ai[0][v] + br);
            float z  = sigmoidf_(ai[1][v] + bz);
            float nn = tanhf_(ai[2][v] + bi_n + r * bh_n);
            out[((size_t)m * Tz + t) * Hz + col] = (1.0f - z) * nn;
        }
    }
}

// Commit the speaker update after all readers of the old state are done.
__global__ void k_commit_spk(const float* __restrict__ spkNew,
                             float* __restrict__ spkStates,
                             const int* __restrict__ sid, int t) {
    int idx = blockIdx.x * blockDim.x + threadIdx.x;  // Bz*Hz = 65536
    int m = idx >> 9, c = idx & (Hz - 1);
    int spk = sid[(size_t)m * Tz + t];
    spkStates[((size_t)m * 2 + spk) * Hz + c] = spkNew[idx];
}

// Wt[k*N3 + n] = bf16(W[n*K + k]); W is [N3 x K] row-major.
__global__ void k_transpose_bf16(const float* __restrict__ W,
                                 __bf16* __restrict__ Wt, int K) {
    int idx = blockIdx.x * blockDim.x + threadIdx.x;
    if (idx >= K * N3) return;
    int n = idx % N3;
    int k = idx / N3;
    Wt[idx] = (__bf16)W[(size_t)n * K + k];
}

extern "C" void kernel_launch(void* const* d_in, const int* in_sizes, int n_in,
                              void* d_out, int out_size, void* d_ws, size_t ws_size,
                              hipStream_t stream) {
    (void)in_sizes; (void)n_in; (void)out_size; (void)ws_size;
    const float* utt  = (const float*)d_in[0];
    const int*   sid  = (const int*)d_in[1];
    const float* gWih = (const float*)d_in[2];
    const float* gWhh = (const float*)d_in[3];
    const float* gbih = (const float*)d_in[4];
    const float* gbhh = (const float*)d_in[5];
    const float* sWih = (const float*)d_in[6];
    const float* sWhh = (const float*)d_in[7];
    const float* sbih = (const float*)d_in[8];
    const float* sbhh = (const float*)d_in[9];
    const float* eWih = (const float*)d_in[10];
    // d_in[11] = eW_hh is mathematically unused (emotion hidden state is zero)
    const float* ebih = (const float*)d_in[12];
    const float* ebhh = (const float*)d_in[13];
    float* out = (float*)d_out;

    char* ws = (char*)d_ws;
    size_t off = 0;
    auto take = [&](size_t bytes) -> char* {
        char* p = ws + off;
        off += (bytes + 255) & ~(size_t)255;
        return p;
    };
    __bf16* WtG_ih = (__bf16*)take((size_t)Dz * N3 * 2);
    __bf16* WtG_hh = (__bf16*)take((size_t)Hz * N3 * 2);
    __bf16* WtS_ih = (__bf16*)take((size_t)(Dz + Hz) * N3 * 2);
    __bf16* WtS_hh = (__bf16*)take((size_t)Hz * N3 * 2);
    __bf16* WtE_ih = (__bf16*)take((size_t)(Dz + Hz) * N3 * 2);
    float* hG0       = (float*)take((size_t)Bz * Hz * 4);
    float* hG1       = (float*)take((size_t)Bz * Hz * 4);
    float* spkStates = (float*)take((size_t)Bz * 2 * Hz * 4);
    float* spkNew    = (float*)take((size_t)Bz * Hz * 4);

    // One-time (per launch) weight transpose + bf16 convert. ~13 MB total ->
    // resident in 192 MB L2 for the whole recurrence.
    auto gridT = [](int K) { return (K * N3 + 255) / 256; };
    k_transpose_bf16<<<gridT(Dz), 256, 0, stream>>>(gWih, WtG_ih, Dz);
    k_transpose_bf16<<<gridT(Hz), 256, 0, stream>>>(gWhh, WtG_hh, Hz);
    k_transpose_bf16<<<gridT(Dz + Hz), 256, 0, stream>>>(sWih, WtS_ih, Dz + Hz);
    k_transpose_bf16<<<gridT(Hz), 256, 0, stream>>>(sWhh, WtS_hh, Hz);
    k_transpose_bf16<<<gridT(Dz + Hz), 256, 0, stream>>>(eWih, WtE_ih, Dz + Hz);

    hipMemsetAsync(hG0, 0, (size_t)Bz * Hz * 4, stream);
    hipMemsetAsync(spkStates, 0, (size_t)Bz * 2 * Hz * 4, stream);

    for (int t = 0; t < Tz; ++t) {
        float* hin  = (t & 1) ? hG1 : hG0;
        float* hout = (t & 1) ? hG0 : hG1;
        k_global_cell<<<32, 128, 0, stream>>>(utt, t, hin, WtG_ih, WtG_hh,
                                              gbih, gbhh, hout);
        k_speaker_cell<<<32, 128, 0, stream>>>(utt, t, hout, sid, spkStates,
                                               WtS_ih, WtS_hh, sbih, sbhh, spkNew);
        k_emotion_cell<<<32, 128, 0, stream>>>(utt, t, spkNew, WtE_ih,
                                               ebih, ebhh, out);
        k_commit_spk<<<256, 256, 0, stream>>>(spkNew, spkStates, sid, t);
    }
}
